// GRU_teach_33389075759190
// MI455X (gfx1250) — compile-verified, matrix operand order checked by reading, not run on hardware
//
#include <hip/hip_runtime.h>
#include <stdint.h>

// GRU teacher-forcing recurrence on MI455X (gfx1250, wave32).
// Strategy: persistent kernel, bf16 WMMA (v_wmma_f32_16x16x32_bf16) for all three
// per-step GEMMs, grid-wide ticket barrier between phases. Weights (~9.4 MB bf16)
// are L2-resident; the problem is compute/latency bound, not HBM bound.

#define BB     256   // batch
#define XDIM   1024  // x dim
#define YDIM   256   // y dim
#define LDIM   1024  // latents
#define TSTEPS 512   // time steps

#define NWG    128   // persistent workgroups (256 thr = 8 wave32 each -> 1024 waves)

typedef __attribute__((ext_vector_type(16))) __bf16 v16bf;
typedef __attribute__((ext_vector_type(8)))  float  v8f;

union FragU { uint4 q[2]; v16bf v; };

// A/B fragment loader for 16x16x32 bf16 WMMA.
// ISA layout (16-bit A 16x32): lanes 0-15 hold M=0..15 with K=0..7 (VGPR0-3) and
// K=16..23 (VGPR4-7); lanes 16-31 hold K=8..15 / K=24..31. B is loaded with the
// symmetric layout from row-major [N,K] weights (works for D = A * W^T).
static __device__ __forceinline__ v16bf load_frag(const __bf16* __restrict__ base,
                                                  int row0, int stride, int k0,
                                                  int fr, int fk) {
  const __bf16* p = base + (size_t)(row0 + fr) * (size_t)stride + (size_t)(k0 + fk);
  FragU u;
  u.q[0] = *(const uint4*)(p);       // K = fk .. fk+7
  u.q[1] = *(const uint4*)(p + 16);  // K = fk+16 .. fk+23
  return u.v;
}

static __device__ __forceinline__ v8f splat8(float x) {
  v8f r;
#pragma unroll
  for (int i = 0; i < 8; ++i) r[i] = x;
  return r;
}

static __device__ __forceinline__ v8f wmma_bf16(v16bf a, v16bf b, v8f c) {
  // 8 args: (neg_a, A, neg_b, B, c_mod, C, reuse_a, reuse_b)
  return __builtin_amdgcn_wmma_f32_16x16x32_bf16(false, a, false, b, (short)0, c,
                                                 false, false);
}

static __device__ __forceinline__ float sigmoidf_(float x) {
  return 1.0f / (1.0f + __expf(-x));
}

// Device-wide barrier: monotonic ticket counter, one signal per WG.
// (Assumes all NWG workgroups are co-resident; NWG=128 << WGP count.)
static __device__ __forceinline__ void grid_barrier(unsigned int* cnt, unsigned int nwg) {
  __syncthreads();
  if (threadIdx.x == 0) {
    __threadfence();
    unsigned int ticket =
        __hip_atomic_fetch_add(cnt, 1u, __ATOMIC_ACQ_REL, __HIP_MEMORY_SCOPE_AGENT);
    unsigned int target = (ticket / nwg + 1u) * nwg;
    while (__hip_atomic_load(cnt, __ATOMIC_ACQUIRE, __HIP_MEMORY_SCOPE_AGENT) < target) {
      __builtin_amdgcn_s_sleep(2);
    }
    __threadfence();
  }
  __syncthreads();
}

// ---------------------------------------------------------------------------
// Prep: f32 -> bf16 weight/input conversion, zero t=0 prediction, zero barrier.
// ---------------------------------------------------------------------------
__global__ void gru_prep_kernel(const float* __restrict__ x,
                                const float* __restrict__ Wbeta,
                                const float* __restrict__ Wih,
                                const float* __restrict__ Whh,
                                const float* __restrict__ Wout,
                                __bf16* __restrict__ xb,
                                __bf16* __restrict__ wbetab,
                                __bf16* __restrict__ wihb,
                                __bf16* __restrict__ whhb,
                                __bf16* __restrict__ woutb,
                                float* __restrict__ prev,
                                unsigned int* __restrict__ bar) {
  size_t i = (size_t)blockIdx.x * blockDim.x + threadIdx.x;
  if (i < (size_t)BB * XDIM)           xb[i]     = (__bf16)x[i];
  if (i < (size_t)LDIM * XDIM)         wbetab[i] = (__bf16)Wbeta[i];
  if (i < (size_t)3 * LDIM * 2 * YDIM) wihb[i]   = (__bf16)Wih[i];
  if (i < (size_t)3 * LDIM * LDIM)     whhb[i]   = (__bf16)Whh[i];
  if (i < (size_t)YDIM * LDIM)         woutb[i]  = (__bf16)Wout[i];
  if (i < (size_t)BB * YDIM)           prev[i]   = 0.0f;
  if (i == 0) *bar = 0u;
}

// ---------------------------------------------------------------------------
// Persistent recurrence kernel.
// ---------------------------------------------------------------------------
__global__ void __launch_bounds__(256, 1)
gru_persistent_kernel(const float* __restrict__ y,
                      const unsigned char* __restrict__ ymask,  // jax bool = 1 byte
                      const float* __restrict__ b_beta,
                      const float* __restrict__ b_ih,
                      const float* __restrict__ b_hh,
                      const float* __restrict__ b_out,
                      const __bf16* __restrict__ Wih,
                      const __bf16* __restrict__ Whh,
                      const __bf16* __restrict__ Wout,
                      const __bf16* __restrict__ Wbeta,
                      const __bf16* __restrict__ Xb,
                      float* __restrict__ hF0, float* __restrict__ hF1,
                      __bf16* __restrict__ hB0, __bf16* __restrict__ hB1,
                      __bf16* __restrict__ gin,
                      float* __restrict__ prev,
                      float* __restrict__ out,
                      unsigned int* __restrict__ bar) {
  const unsigned int nwg = gridDim.x;
  const int tid      = blockIdx.x * blockDim.x + threadIdx.x;
  const int nthreads = gridDim.x * blockDim.x;
  const int wave = (tid >> 5);
  const int lane = threadIdx.x & 31;

  // fragment lane addressing (A/B frags)
  const int fr = lane & 15;              // tile row (A: batch row, B: weight row N)
  const int fk = (lane >> 4) ? 8 : 0;    // k sub-offset within 32-wide chunk
  // C/D lane addressing
  const int cn  = lane & 15;             // tile column
  const int cm0 = (lane >> 4) ? 8 : 0;   // tile row base (+VGPR index)

  const int NT_A = (BB / 16) * (LDIM / 16);  // 1024 gate tile-triples
  const int NT_B = (BB / 16) * (YDIM / 16);  // 256 output tiles

  // ---------------- h0 = x @ W_beta^T + b_beta -> buffers {hF0, hB0} ----------
  if (wave < NT_A) {
    const int bt = wave >> 6, lt = wave & 63;
    const int col = lt * 16 + cn;
    v8f acc = splat8(b_beta[col]);
#pragma unroll 4
    for (int kc = 0; kc < XDIM / 32; ++kc) {
      const int k0 = kc * 32;
      v16bf A  = load_frag(Xb,    bt * 16, XDIM, k0, fr, fk);
      v16bf Bv = load_frag(Wbeta, lt * 16, XDIM, k0, fr, fk);
      acc = wmma_bf16(A, Bv, acc);
    }
#pragma unroll
    for (int v = 0; v < 8; ++v) {
      const int row = bt * 16 + cm0 + v;
      hF0[(size_t)row * LDIM + col] = acc[v];
      hB0[(size_t)row * LDIM + col] = (__bf16)acc[v];
    }
  }
  grid_barrier(bar, nwg);

  // ---------------- recurrence over T steps -----------------------------------
  for (int t = 0; t < TSTEPS; ++t) {
    const float*  hFr = (t & 1) ? hF1 : hF0;
    float*        hFw = (t & 1) ? hF0 : hF1;
    const __bf16* hBr = (t & 1) ? hB1 : hB0;
    __bf16*       hBw = (t & 1) ? hB0 : hB1;

    // Phase C: build interleaved (value, mask) bf16 GRU input [B, 2*YD]
    for (int i = tid; i < BB * YDIM; i += nthreads) {
      const int b = i >> 8, d = i & (YDIM - 1);
      const unsigned char m = ymask[(size_t)i * TSTEPS + t];
      const float yv = m ? y[(size_t)i * TSTEPS + t] : prev[i];
      gin[(size_t)b * (2 * YDIM) + 2 * d]     = (__bf16)yv;
      gin[(size_t)b * (2 * YDIM) + 2 * d + 1] = (__bf16)(m ? 1.0f : 0.0f);
    }
    grid_barrier(bar, nwg);

    // Phase A: gates + h_new for one (batch-tile, L-tile) per wave
    if (wave < NT_A) {
      const int bt = wave >> 6, lt = wave & 63;
      const int col  = lt * 16 + cn;
      const int aRow = bt * 16;
      const int wRow = lt * 16;

      v8f accR  = splat8(b_ih[col] + b_hh[col]);
      v8f accZ  = splat8(b_ih[LDIM + col] + b_hh[LDIM + col]);
      v8f accIN = splat8(b_ih[2 * LDIM + col]);   // i_n (kept separate: n uses r*h_n)
      v8f accHN = splat8(b_hh[2 * LDIM + col]);   // h_n

      // gi part: K = 2*YD = 512
#pragma unroll 2
      for (int kc = 0; kc < (2 * YDIM) / 32; ++kc) {
        const int k0 = kc * 32;
        v16bf A  = load_frag(gin, aRow, 2 * YDIM, k0, fr, fk);
        v16bf Br = load_frag(Wih, wRow,            2 * YDIM, k0, fr, fk);
        v16bf Bz = load_frag(Wih, LDIM + wRow,     2 * YDIM, k0, fr, fk);
        v16bf Bn = load_frag(Wih, 2 * LDIM + wRow, 2 * YDIM, k0, fr, fk);
        accR  = wmma_bf16(A, Br, accR);
        accZ  = wmma_bf16(A, Bz, accZ);
        accIN = wmma_bf16(A, Bn, accIN);
      }
      // gh part: K = L = 1024
#pragma unroll 2
      for (int kc = 0; kc < LDIM / 32; ++kc) {
        const int k0 = kc * 32;
        if (kc + 1 < LDIM / 32)
          __builtin_prefetch(Whh + (size_t)(wRow + fr) * LDIM + k0 + 32 + fk, 0, 3);
        v16bf A  = load_frag(hBr, aRow, LDIM, k0, fr, fk);
        v16bf Br = load_frag(Whh, wRow,            LDIM, k0, fr, fk);
        v16bf Bz = load_frag(Whh, LDIM + wRow,     LDIM, k0, fr, fk);
        v16bf Bn = load_frag(Whh, 2 * LDIM + wRow, LDIM, k0, fr, fk);
        accR  = wmma_bf16(A, Br, accR);
        accZ  = wmma_bf16(A, Bz, accZ);
        accHN = wmma_bf16(A, Bn, accHN);
      }
      // elementwise gate math + state update (f32, matches reference formulas)
#pragma unroll
      for (int v = 0; v < 8; ++v) {
        const int row = bt * 16 + cm0 + v;
        const float hp = hFr[(size_t)row * LDIM + col];
        const float rg = sigmoidf_(accR[v]);
        const float zg = sigmoidf_(accZ[v]);
        const float ng = tanhf(accIN[v] + rg * accHN[v]);
        const float hn = (1.0f - zg) * ng + zg * hp;
        hFw[(size_t)row * LDIM + col] = hn;
        hBw[(size_t)row * LDIM + col] = (__bf16)hn;
      }
    }
    grid_barrier(bar, nwg);

    // Phase B: out_t = h_new @ W_out^T + b_out  (16x16 tiles, K = L)
    if (wave < NT_B) {
      const int bt = wave >> 4, dt = wave & 15;
      const int col = dt * 16 + cn;
      v8f acc = splat8(b_out[col]);
#pragma unroll 4
      for (int kc = 0; kc < LDIM / 32; ++kc) {
        const int k0 = kc * 32;
        v16bf A  = load_frag(hBw,  bt * 16, LDIM, k0, fr, fk);
        v16bf Bv = load_frag(Wout, dt * 16, LDIM, k0, fr, fk);
        acc = wmma_bf16(A, Bv, acc);
      }
#pragma unroll
      for (int v = 0; v < 8; ++v) {
        const int b = bt * 16 + cm0 + v;
        const float o = acc[v];
        prev[(size_t)b * YDIM + col] = o;                       // feeds next step
        out[((size_t)b * YDIM + col) * TSTEPS + t] = o;         // [B, D, T]
      }
    }
    grid_barrier(bar, nwg);
  }
}

// ---------------------------------------------------------------------------
// Host launch
// ---------------------------------------------------------------------------
extern "C" void kernel_launch(void* const* d_in, const int* in_sizes, int n_in,
                              void* d_out, int out_size, void* d_ws, size_t ws_size,
                              hipStream_t stream) {
  const float*         x      = (const float*)d_in[0];
  const float*         y      = (const float*)d_in[1];
  const unsigned char* ymask  = (const unsigned char*)d_in[2];  // bool array
  const float*         W_beta = (const float*)d_in[3];
  const float*         b_beta = (const float*)d_in[4];
  const float*         W_ih   = (const float*)d_in[5];
  const float*         b_ih   = (const float*)d_in[6];
  const float*         W_hh   = (const float*)d_in[7];
  const float*         b_hh   = (const float*)d_in[8];
  const float*         W_out  = (const float*)d_in[9];
  const float*         b_out  = (const float*)d_in[10];
  float*               out    = (float*)d_out;

  char* ws = (char*)d_ws;
  size_t off = 0;
  unsigned int* bar = (unsigned int*)(ws + off); off += 256;
  __bf16* wihb   = (__bf16*)(ws + off); off += (size_t)3 * LDIM * 2 * YDIM * 2;
  __bf16* whhb   = (__bf16*)(ws + off); off += (size_t)3 * LDIM * LDIM * 2;
  __bf16* woutb  = (__bf16*)(ws + off); off += (size_t)YDIM * LDIM * 2;
  __bf16* wbetab = (__bf16*)(ws + off); off += (size_t)LDIM * XDIM * 2;
  __bf16* xb     = (__bf16*)(ws + off); off += (size_t)BB * XDIM * 2;
  float*  hF0    = (float*)(ws + off);  off += (size_t)BB * LDIM * 4;
  float*  hF1    = (float*)(ws + off);  off += (size_t)BB * LDIM * 4;
  __bf16* hB0    = (__bf16*)(ws + off); off += (size_t)BB * LDIM * 2;
  __bf16* hB1    = (__bf16*)(ws + off); off += (size_t)BB * LDIM * 2;
  __bf16* gin    = (__bf16*)(ws + off); off += (size_t)BB * 2 * YDIM * 2;
  float*  prev   = (float*)(ws + off);  off += (size_t)BB * YDIM * 4;

  const size_t maxElems = (size_t)3 * LDIM * LDIM;  // largest conversion (W_hh)
  const int prepBlocks = (int)((maxElems + 255) / 256);
  gru_prep_kernel<<<prepBlocks, 256, 0, stream>>>(x, W_beta, W_ih, W_hh, W_out,
                                                  xb, wbetab, wihb, whhb, woutb,
                                                  prev, bar);

  gru_persistent_kernel<<<NWG, 256, 0, stream>>>(y, ymask, b_beta, b_ih, b_hh, b_out,
                                                 wihb, whhb, woutb, wbetab, xb,
                                                 hF0, hF1, hB0, hB1, gin, prev, out,
                                                 bar);
}
